// FedFormer_40372692582804
// MI455X (gfx1250) — matrix operand order repaired and checked by reading, not compiled, not run on hardware
//
#include <hip/hip_runtime.h>
#include <math.h>

typedef __attribute__((ext_vector_type(16))) _Float16 v16h;
typedef __attribute__((ext_vector_type(8)))  float    v8f;
typedef unsigned int u32x4 __attribute__((ext_vector_type(4)));
typedef int          i32x8 __attribute__((ext_vector_type(8)));
typedef int          i32x4 __attribute__((ext_vector_type(4)));

// ---------------- problem constants ----------------
#define FB    64      // batch
#define FSEQ  1024
#define FPRED 512
#define FCIN  7
#define FD    64
#define FH    8
#define FE    8
#define FDFF  256
#define FMODE 64
#define FTOK  (FB*FSEQ)     // 65536 tokens
#define FFR   (FB*FH*FE)    // 4096 frequency rows

// =====================================================================
// Generic WMMA GEMM:  C[M,N] = act( A[M,K] @ W[N,K]^T + bias )
//   A addressing:  &A[(m/rowDiv)*rowOuter + (m%rowDiv)*rowInner + k*strideK]
//   storeMode 0 : C[m*ldc + n]
//   storeMode 1 : fourier "faithful view" scramble (m=(b,h,e) row, n=l)
// Block = 256 threads (8 waves); block tile = 128 rows x NB cols.
// A tile staged to LDS as f16 by all threads (handles strided rows).
// W tile staged to LDS as f32 by the Tensor Data Mover (2D D# descriptor,
// LDS pad of 1 DWORD per 32 DWORDs -> row stride 33 floats), fenced with
// s_wait_tensorcnt + workgroup barrier.  Cooperative-load fallback when
// the TDM builtin is unavailable (host pass / older toolchains).
// =====================================================================
template<int NB>
__global__ __launch_bounds__(256)
void wmma_gemm_kernel(const float* __restrict__ A, const float* __restrict__ W,
                      const float* __restrict__ bias, float* __restrict__ C,
                      int M, int K, int N,
                      int rowDiv, long rowOuter, long rowInner, long strideK,
                      long ldc, int storeMode, int doGelu)
{
    __shared__ _Float16 sA[128][36];
    __shared__ float    sBf[NB][33];     // matches TDM pad: 32 data + 1 pad DW

    const int tid  = threadIdx.x;
    const int wave = tid >> 5;
    const int lane = tid & 31;
    const long mBase = (long)blockIdx.x * 128;
    const long nBase = (long)blockIdx.y * NB;

    v8f acc[NB / 16];
#pragma unroll
    for (int t = 0; t < NB / 16; ++t)
#pragma unroll
        for (int q = 0; q < 8; ++q) acc[t][q] = 0.0f;

    for (int k0 = 0; k0 < K; k0 += 32) {
        // ---- stage A tile (128 x 32) to LDS as f16 (handles strides) ----
        for (int e = tid; e < 128 * 32; e += 256) {
            int r = e >> 5, c = e & 31;
            long m = mBase + r;
            long base = (m / rowDiv) * rowOuter + (m % rowDiv) * rowInner;
            const float* ap = A + base + (long)(k0 + c) * strideK;
            sA[r][c] = (_Float16)(*ap);
            if (k0 + 32 < K) __builtin_prefetch(ap + 32 * strideK, 0, 0);
        }
        // ---- stage W tile (NB x 32) to LDS via Tensor Data Mover ----
#if defined(__gfx1250__) && __has_builtin(__builtin_amdgcn_tensor_load_to_lds)
        if (wave == 0) {
            unsigned long long ga =
                (unsigned long long)(const void*)(W + nBase * (long)K + k0);
            unsigned ldsB = (unsigned)(unsigned long long)(const void*)&sBf[0][0];
            u32x4 g0;
            g0[0] = 1u;                                    // count=1 (valid D#)
            g0[1] = ldsB;                                  // lds_addr (bytes)
            g0[2] = (unsigned)(ga & 0xffffffffull);        // global_addr[31:0]
            g0[3] = (unsigned)((ga >> 32) & 0x1ffffffull)  // global_addr[56:32]
                  | (2u << 30);                            // type=2 ("image")
            i32x8 g1;
            g1[0] = (2 << 16)    // data_size = 4 bytes
                  | (1 << 20)    // pad_enable
                  | (4 << 22);   // pad_interval: every 32 DWORDs, +1 DWORD
            g1[1] = (int)(32u << 16);             // tensor_dim0 = 32 (lo16)
            g1[2] = (int)((unsigned)NB << 16);    // tensor_dim1 = NB (lo16)
            g1[3] = (int)(32u << 16);             // tile_dim0 = 32
            g1[4] = NB;                           // tile_dim1 = NB
            g1[5] = K;                            // tensor_dim0_stride = K
            g1[6] = 0;
            g1[7] = 0;
            i32x4 gz; gz[0] = 0; gz[1] = 0; gz[2] = 0; gz[3] = 0;
#if __clang_major__ >= 23
            i32x8 gz8;
            for (int q = 0; q < 8; ++q) gz8[q] = 0;
            __builtin_amdgcn_tensor_load_to_lds(g0, g1, gz, gz, gz8, 0);
#else
            __builtin_amdgcn_tensor_load_to_lds(g0, g1, gz, gz, 0);
#endif
            __builtin_amdgcn_s_wait_tensorcnt(0);
        }
#else
        for (int e = tid; e < NB * 32; e += 256) {
            int r = e >> 5, c = e & 31;
            sBf[r][c] = W[(nBase + r) * (long)K + k0 + c];
        }
#endif
        __syncthreads();

        // ---- A fragment: 16x32 f16 layout (wave32) ----
        v16h af;
        {
            const int am   = lane & 15;
            const int kb   = (lane >> 4) * 8;
            const int mrow = wave * 16 + am;
#pragma unroll
            for (int j = 0; j < 8; ++j) {
                int k = (j < 4) ? (kb + 2 * j) : (16 + kb + 2 * (j - 4));
                af[2 * j]     = sA[mrow][k];
                af[2 * j + 1] = sA[mrow][k + 1];
            }
        }
        // ---- per 16-col tile: B fragment (32x16, f32->f16) + WMMA ----
#pragma unroll
        for (int t = 0; t < NB / 16; ++t) {
            v16h bf;
            const int bn  = t * 16 + (lane & 15);
            const int kbB = (lane >> 4) * 8;
#pragma unroll
            for (int j = 0; j < 8; ++j) {
                int k = (j < 4) ? (kbB + 2 * j) : (16 + kbB + 2 * (j - 4));
                bf[2 * j]     = (_Float16)sBf[bn][k];
                bf[2 * j + 1] = (_Float16)sBf[bn][k + 1];
            }
            acc[t] = __builtin_amdgcn_wmma_f32_16x16x32_f16(
                false, af, false, bf, (short)0, acc[t], false, false);
        }
        __syncthreads();
    }

    // ---- epilogue: C/D layout -> global ----
    const int nLoc = lane & 15;
    const int mOff = (lane >> 4) * 8;
#pragma unroll
    for (int t = 0; t < NB / 16; ++t) {
        long n  = nBase + t * 16 + nLoc;
        float bv = bias ? bias[n] : 0.0f;
#pragma unroll
        for (int r = 0; r < 8; ++r) {
            long m = mBase + wave * 16 + mOff + r;
            float v = acc[t][r] + bv;
            if (doGelu) v = 0.5f * v * (1.0f + erff(v * 0.70710678118f));
            if (storeMode == 0) {
                C[m * ldc + n] = v;
            } else {                      // faithful (B,H,E,L)->(B,L,64) view
                long b    = m >> 6;
                long flat = (m & 63) * 1024 + n;
                C[b * 65536 + flat] = v;
            }
        }
    }
}

// =====================================================================
// series_decomp_multi via LDS prefix scan. One block = one (b,c) column.
// =====================================================================
__global__ __launch_bounds__(1024)
void decomp_kernel(const float* __restrict__ X, const float* __restrict__ X2,
                   int C, int L, long bStr, long cStr, long lStr,
                   const float* __restrict__ w4, const float* __restrict__ b4,
                   float* __restrict__ seasonalOut, float* __restrict__ meanOut,
                   int meanAcc, float* __restrict__ seqMeanOut)
{
    __shared__ float xs[1024];
    __shared__ float pA[1024];
    __shared__ float pB[1024];

    const int bc = blockIdx.x;
    const int b = bc / C, c = bc % C;
    const int i = threadIdx.x;
    const long base = (long)b * bStr + (long)c * cStr;

    float xv = X[base + (long)i * lStr];
    if (X2) xv += X2[base + (long)i * lStr];
    xs[i] = xv;
    pA[i] = xv;
    __syncthreads();

    float* src = pA;
    float* dst = pB;
    for (int off = 1; off < 1024; off <<= 1) {
        float v = src[i];
        if (i >= off) v += src[i - off];
        dst[i] = v;
        __syncthreads();
        float* tmp = src; src = dst; dst = tmp;
    }
    const float* P = src;              // inclusive prefix sum
    const float x0 = xs[0], xe = xs[L - 1];

    const int ks[4] = {10, 50, 100, 500};
    float num = 0.0f, den = 0.0f;
#pragma unroll
    for (int kk = 0; kk < 4; ++kk) {
        int k  = ks[kk];
        int lo = i - (k - 1) / 2, hi = i + k / 2;
        int nlo = lo < 0 ? -lo : 0;
        int nhi = hi > L - 1 ? hi - (L - 1) : 0;
        int loc = lo < 0 ? 0 : lo;
        int hic = hi > L - 1 ? L - 1 : hi;
        float s = P[hic] - (loc > 0 ? P[loc - 1] : 0.0f)
                + (float)nlo * x0 + (float)nhi * xe;
        float avg = s / (float)k;
        float e = __expf(xv * w4[kk] + b4[kk]);
        den += e;
        num += avg * e;
    }
    float mean = num / den;
    long o = ((long)b * L + i) * C + c;
    seasonalOut[o] = xv - mean;
    if (meanOut) {
        if (meanAcc) meanOut[o] += mean; else meanOut[o] = mean;
    }
    if (seqMeanOut && i == 0) seqMeanOut[b * C + c] = P[L - 1] / (float)L;
}

// ---- DataEmbedding_onlypos: circular conv1d(k=3, Cin=7 -> 64) + sin/cos PE
__global__ void embed_kernel(const float* __restrict__ X,
                             long bStr, long cStr, long lStr,
                             const float* __restrict__ Wc,   // (64,7,3)
                             float* __restrict__ out, int L)
{
    long idx = (long)blockIdx.x * 256 + threadIdx.x;     // B*L*64
    int d = (int)(idx & 63);
    long bl = idx >> 6;
    int l = (int)(bl % L);
    int b = (int)(bl / L);
    float s = 0.0f;
#pragma unroll
    for (int c = 0; c < FCIN; ++c)
#pragma unroll
        for (int t = 0; t < 3; ++t) {
            int ll = l - 1 + t;
            ll = (ll + L) % L;
            s += X[(long)b * bStr + (long)c * cStr + (long)ll * lStr]
               * Wc[(d * FCIN + c) * 3 + t];
        }
    int j = d >> 1;
    float dv  = __expf((float)(2 * j) * (-9.210340371976184f / 64.0f));
    float ang = (float)l * dv;
    s += (d & 1) ? __cosf(ang) : __sinf(ang);
    out[idx] = s;
}

// ---- complex per-head mode mixing: OW[(b,h,o)][m / 64+m] = sum_e QF * (wr+i wi)
__global__ void mode_mix_kernel(const float* __restrict__ QF,
                                const float* __restrict__ wr,
                                const float* __restrict__ wi,
                                float* __restrict__ OW, float scale)
{
    int idx = blockIdx.x * 256 + threadIdx.x;  // B*H*E*M = 32768
    int m = idx & 63;
    int o = (idx >> 6) & 7;
    int h = (idx >> 9) & 7;
    int b = idx >> 12;
    float sr = 0.0f, si = 0.0f;
#pragma unroll
    for (int e = 0; e < FE; ++e) {
        long row = (long)(b * 64 + h * 8 + e) * 128;
        float xr = QF[row + m], xi = QF[row + 64 + m];
        long wo = (((long)(h * 8 + e) * 8 + o) * 64) + m;
        float wR = wr[wo], wI = wi[wo];
        sr += xr * wR - xi * wI;
        si += xr * wI + xi * wR;
    }
    long orow = (long)(b * 64 + h * 8 + o) * 128;
    OW[orow + m]      = sr * scale;
    OW[orow + 64 + m] = si * scale;
}

// ---- cross attention: qk = tanh_c( sum_e qf*kf ) ----
__global__ void cross_qk_kernel(const float* __restrict__ QF,
                                const float* __restrict__ KF,
                                float* __restrict__ QKr, float* __restrict__ QKi)
{
    int idx = blockIdx.x * 256 + threadIdx.x;  // B*H*64*64
    int y = idx & 63;
    int x = (idx >> 6) & 63;
    int h = (idx >> 12) & 7;
    int b = idx >> 15;
    float sr = 0.0f, si = 0.0f;
#pragma unroll
    for (int e = 0; e < FE; ++e) {
        long row = (long)(b * 64 + h * 8 + e) * 128;
        float qr = QF[row + x], qi = QF[row + 64 + x];
        float kr = KF[row + y], ki = KF[row + 64 + y];
        sr += qr * kr - qi * ki;
        si += qr * ki + qi * kr;
    }
    // complex tanh(a+ib) = (sinh 2a + i sin 2b)/(cosh 2a + cos 2b)
    float s2a = sinhf(2.0f * sr), c2a = coshf(2.0f * sr);
    float s2b = __sinf(2.0f * si), c2b = __cosf(2.0f * si);
    float den = c2a + c2b;
    QKr[idx] = s2a / den;
    QKi[idx] = s2b / den;
}

// ---- qkv[(b,h,e)][x] = sum_y qk[x,y] * kf[e,y] (complex) ----
__global__ void cross_qkv_kernel(const float* __restrict__ QKr,
                                 const float* __restrict__ QKi,
                                 const float* __restrict__ KF,
                                 float* __restrict__ QKV)
{
    int idx = blockIdx.x * 256 + threadIdx.x;  // B*H*E*64 = 32768
    int x = idx & 63;
    int e = (idx >> 6) & 7;
    int h = (idx >> 9) & 7;
    int b = idx >> 12;
    long qkBase = ((long)(b * 8 + h) * 64 + x) * 64;
    long krow   = (long)(b * 64 + h * 8 + e) * 128;
    float sr = 0.0f, si = 0.0f;
    for (int y = 0; y < 64; ++y) {
        float ar = QKr[qkBase + y], ai = QKi[qkBase + y];
        float kr = KF[krow + y],    ki = KF[krow + 64 + y];
        sr += ar * kr - ai * ki;
        si += ar * ki + ai * kr;
    }
    long orow = (long)(b * 64 + h * 8 + e) * 128;
    QKV[orow + x]      = sr;
    QKV[orow + 64 + x] = si;
}

// ---- my_Layernorm pass 1: LN over D per token ----
__global__ void myln_pass1(const float* __restrict__ X, const float* __restrict__ g,
                           const float* __restrict__ bt, float* __restrict__ out)
{
    long t = (long)blockIdx.x * 256 + threadIdx.x;  // B*L
    const float* x = X + t * FD;
    float mu = 0.0f;
    for (int d = 0; d < FD; ++d) mu += x[d];
    mu *= (1.0f / FD);
    float var = 0.0f;
    for (int d = 0; d < FD; ++d) { float dv = x[d] - mu; var += dv * dv; }
    var *= (1.0f / FD);
    float inv = rsqrtf(var + 1e-5f);
    for (int d = 0; d < FD; ++d)
        out[t * FD + d] = (x[d] - mu) * inv * g[d] + bt[d];
}

// ---- my_Layernorm pass 2: subtract per-(b,d) sequence mean, in place ----
__global__ void colmean_sub_kernel(float* __restrict__ X, int L)
{
    __shared__ float red[256];
    int bd = blockIdx.x;
    int b = bd >> 6, d = bd & 63;
    float* base = X + (long)b * L * FD + d;
    float s = 0.0f;
    for (int l = threadIdx.x; l < L; l += 256) s += base[(long)l * FD];
    red[threadIdx.x] = s;
    __syncthreads();
    for (int off = 128; off; off >>= 1) {
        if (threadIdx.x < off) red[threadIdx.x] += red[threadIdx.x + off];
        __syncthreads();
    }
    float mu = red[0] / (float)L;
    for (int l = threadIdx.x; l < L; l += 256) base[(long)l * FD] -= mu;
}

// ---- trend_init projection: concat(trend[-512:], mean) @ dp_w + dp_b ----
__global__ void trend_base_kernel(const float* __restrict__ trend7,
                                  const float* __restrict__ mean7,
                                  const float* __restrict__ dpw,
                                  const float* __restrict__ dpb,
                                  float* __restrict__ out)
{
    int idx = blockIdx.x * 256 + threadIdx.x;   // B*1024
    int l = idx & 1023, b = idx >> 10;
    float s = dpb[0];
#pragma unroll
    for (int c = 0; c < FCIN; ++c) {
        float v = (l < 512) ? trend7[((long)b * 1024 + 512 + l) * FCIN + c]
                            : mean7[b * FCIN + c];
        s += v * dpw[c];
    }
    out[idx] = s;
}

// ---- seasonal_init: pad(seasonal[-512:], PRED zeros) ----
__global__ void seasonal_init_kernel(const float* __restrict__ s7,
                                     float* __restrict__ out)
{
    long idx = (long)blockIdx.x * 256 + threadIdx.x;  // B*1024*7
    int c = (int)(idx % FCIN);
    long bl = idx / FCIN;
    int l = (int)(bl % 1024);
    int b = (int)(bl / 1024);
    out[idx] = (l < 512) ? s7[((long)b * 1024 + 512 + l) * FCIN + c] : 0.0f;
}

// ---- trend = trend_base + circ_conv(t1+t2+t3, dec_trend_w) ----
__global__ void trend_conv_kernel(const float* __restrict__ tsum,
                                  const float* __restrict__ tw,   // (1,64,3)
                                  const float* __restrict__ tbase,
                                  float* __restrict__ out)
{
    int idx = blockIdx.x * 256 + threadIdx.x;   // B*1024
    int l = idx & 1023, b = idx >> 10;
    float s = tbase[idx];
    for (int d = 0; d < FD; ++d)
#pragma unroll
        for (int t = 0; t < 3; ++t) {
            int ll = (l - 1 + t + 1024) & 1023;
            s += tsum[((long)b * 1024 + ll) * FD + d] * tw[d * 3 + t];
        }
    out[idx] = s;
}

// ---- final: out[:, -PRED:, 0] = xd @ proj_w + proj_b + trend ----
__global__ void final_out_kernel(const float* __restrict__ xd,
                                 const float* __restrict__ pw,
                                 const float* __restrict__ pb,
                                 const float* __restrict__ trend,
                                 float* __restrict__ out)
{
    int idx = blockIdx.x * 256 + threadIdx.x;   // B*512
    int p = idx & 511, b = idx >> 9;
    int l = 512 + p;
    const float* x = xd + ((long)b * 1024 + l) * FD;
    float s = pb[0];
    for (int d = 0; d < FD; ++d) s += x[d] * pw[d];
    out[idx] = s + trend[b * 1024 + l];
}

// ---- DFT bases: Gf[N=128][K=1024] forward, Gi[N=1024][K=128] inverse ----
__global__ void basis_init_kernel(float* __restrict__ Gf, float* __restrict__ Gi)
{
    int idx = blockIdx.x * 256 + threadIdx.x;   // 131072
    {
        int l = idx & 1023, n = idx >> 10;      // Gf row n, col l
        int m = n & 63;
        int lm = (l * m) & 1023;
        float th = 6.283185307179586f * (float)lm / 1024.0f;
        Gf[idx] = (n < 64) ? __cosf(th) : -__sinf(th);
    }
    {
        int k = idx & 127, l = idx >> 7;        // Gi row l, col k
        int m = k & 63;
        int lm = (l * m) & 1023;
        float th = 6.283185307179586f * (float)lm / 1024.0f;
        float sc = (m == 0 ? 1.0f : 2.0f) / 1024.0f;
        Gi[idx] = ((k < 64) ? __cosf(th) : -__sinf(th)) * sc;
    }
}

// =====================================================================
// Host-side launch helpers
// =====================================================================
static inline void gemm_tok64(hipStream_t s, const float* A, int K,
                              const float* W, const float* bias, float* C,
                              int gelu)
{
    wmma_gemm_kernel<64><<<dim3(FTOK / 128, 1), 256, 0, s>>>(
        A, W, bias, C, FTOK, K, 64, 1, (long)K, 0, 1, 64, 0, gelu);
}
static inline void gemm_rfft(hipStream_t s, const float* A, const float* Gf,
                             float* QF)
{   // rows (b,d): base=(row/64)*65536 + (row%64), strideK(l)=64
    wmma_gemm_kernel<128><<<dim3(FFR / 128, 1), 256, 0, s>>>(
        A, Gf, nullptr, QF, FFR, 1024, 128, 64, 65536, 1, 64, 128, 0, 0);
}
static inline void gemm_irfft(hipStream_t s, const float* OW, const float* Gi,
                              float* Y)
{   // rows contiguous [4096][128]; scramble store to (B,L,64)
    wmma_gemm_kernel<128><<<dim3(FFR / 128, 8), 256, 0, s>>>(
        OW, Gi, nullptr, Y, FFR, 128, 1024, 1, 128, 0, 1, 0, 1, 0);
}

extern "C" void kernel_launch(void* const* d_in, const int* in_sizes, int n_in,
                              void* d_out, int out_size, void* d_ws, size_t ws_size,
                              hipStream_t stream)
{
    (void)in_sizes; (void)n_in; (void)out_size; (void)ws_size;
    const float* x_enc       = (const float*)d_in[0];
    const float* dp_w        = (const float*)d_in[1];
    const float* dp_b        = (const float*)d_in[2];
    const float* emb_enc_w   = (const float*)d_in[3];
    const float* emb_dec_w   = (const float*)d_in[4];
    const float* decomp_w    = (const float*)d_in[5];
    const float* decomp_b    = (const float*)d_in[6];
    const float* enc_attn_w  = (const float*)d_in[7];
    const float* enc_attn_b  = (const float*)d_in[8];
    const float* enc_four_wr = (const float*)d_in[9];
    const float* enc_four_wi = (const float*)d_in[10];
    const float* enc_c1      = (const float*)d_in[11];
    const float* enc_c2      = (const float*)d_in[12];
    const float* enc_ln_g    = (const float*)d_in[13];
    const float* enc_ln_b    = (const float*)d_in[14];
    const float* dec_self_w  = (const float*)d_in[15];
    const float* dec_self_b  = (const float*)d_in[16];
    const float* dec_four_wr = (const float*)d_in[17];
    const float* dec_four_wi = (const float*)d_in[18];
    const float* dec_cross_w = (const float*)d_in[19];
    const float* dec_cross_b = (const float*)d_in[20];
    const float* dec_cross_wr= (const float*)d_in[21];
    const float* dec_cross_wi= (const float*)d_in[22];
    const float* dec_c1      = (const float*)d_in[23];
    const float* dec_c2      = (const float*)d_in[24];
    const float* dec_trend_w = (const float*)d_in[25];
    const float* dec_ln_g    = (const float*)d_in[26];
    const float* dec_ln_b    = (const float*)d_in[27];
    const float* dec_proj_w  = (const float*)d_in[28];
    const float* dec_proj_b  = (const float*)d_in[29];
    float* out = (float*)d_out;

    // ---- workspace bump allocator (floats) ----
    float* ws = (float*)d_ws;
    size_t off = 0;
    auto alloc = [&](size_t n) { float* p = ws + off; off += n; return p; };
    float* Gf        = alloc(128 * 1024);
    float* Gi        = alloc(1024 * 128);
    float* seasonal7 = alloc((size_t)FB * 1024 * FCIN);
    float* trend7    = alloc((size_t)FB * 1024 * FCIN);
    float* mean7     = alloc(FB * FCIN);
    float* trendBase = alloc(FB * 1024);
    float* seasInit  = alloc((size_t)FB * 1024 * FCIN);
    float* Hbuf      = alloc((size_t)FTOK * FD);
    float* T1        = alloc((size_t)FTOK * FD);
    float* T2        = alloc((size_t)FTOK * FD);
    float* FFNb      = alloc((size_t)FTOK * FDFF);
    float* HENC      = alloc((size_t)FTOK * FD);
    float* XD        = alloc((size_t)FTOK * FD);
    float* TSUM      = alloc((size_t)FTOK * FD);
    float* QF        = alloc((size_t)FFR * 128);
    float* KF        = alloc((size_t)FFR * 128);
    float* OW        = alloc((size_t)FFR * 128);
    float* QKr       = alloc((size_t)FB * FH * 64 * 64);
    float* QKi       = alloc((size_t)FB * FH * 64 * 64);
    float* QKV       = alloc((size_t)FFR * 128);
    float* trendFin  = alloc(FB * 1024);

    basis_init_kernel<<<131072 / 256, 256, 0, stream>>>(Gf, Gi);

    // ---- input decomposition (x_enc is (B,7,L): stride pattern) ----
    decomp_kernel<<<FB * FCIN, 1024, 0, stream>>>(
        x_enc, nullptr, FCIN, 1024, (long)FCIN * 1024, 1024, 1,
        decomp_w + 0, decomp_b + 0, seasonal7, trend7, 0, mean7);
    trend_base_kernel<<<FB * 1024 / 256, 256, 0, stream>>>(
        trend7, mean7, dp_w, dp_b, trendBase);
    seasonal_init_kernel<<<(FB * 1024 * FCIN) / 256, 256, 0, stream>>>(
        seasonal7, seasInit);

    // ---- encoder embedding ----
    embed_kernel<<<(FTOK * FD) / 256, 256, 0, stream>>>(
        x_enc, (long)FCIN * 1024, 1024, 1, emb_enc_w, Hbuf, 1024);

    int di = 1;
    for (int l = 0; l < 2; ++l) {
        const float* Wq = enc_attn_w + (l * 4 + 0) * 4096;
        const float* bq = enc_attn_b + (l * 4 + 0) * 64;
        const float* Wo = enc_attn_w + (l * 4 + 3) * 4096;
        const float* bo = enc_attn_b + (l * 4 + 3) * 64;
        gemm_tok64(stream, Hbuf, 64, Wq, bq, T1, 0);
        gemm_rfft(stream, T1, Gf, QF);
        mode_mix_kernel<<<32768 / 256, 256, 0, stream>>>(
            QF, enc_four_wr + l * 32768, enc_four_wi + l * 32768, OW, 1.0f);
        gemm_irfft(stream, OW, Gi, T2);
        gemm_tok64(stream, T2, 64, Wo, bo, T1, 0);
        decomp_kernel<<<FB * FD, 1024, 0, stream>>>(
            Hbuf, T1, FD, 1024, (long)1024 * FD, 1, FD,
            decomp_w + di * 4, decomp_b + di * 4, Hbuf, nullptr, 0, nullptr);
        // FFN (no bias): gelu(h@c1^T)@c2^T
        wmma_gemm_kernel<128><<<dim3(FTOK / 128, 2), 256, 0, stream>>>(
            Hbuf, enc_c1 + l * 16384, nullptr, FFNb,
            FTOK, 64, 256, 1, 64, 0, 1, 256, 0, 1);
        gemm_tok64(stream, FFNb, 256, enc_c2 + l * 16384, nullptr, T1, 0);
        decomp_kernel<<<FB * FD, 1024, 0, stream>>>(
            Hbuf, T1, FD, 1024, (long)1024 * FD, 1, FD,
            decomp_w + (di + 1) * 4, decomp_b + (di + 1) * 4,
            Hbuf, nullptr, 0, nullptr);
        di += 2;
    }
    myln_pass1<<<FTOK / 256, 256, 0, stream>>>(Hbuf, enc_ln_g, enc_ln_b, HENC);
    colmean_sub_kernel<<<FB * FD, 256, 0, stream>>>(HENC, 1024);

    // ---- decoder embedding ----
    embed_kernel<<<(FTOK * FD) / 256, 256, 0, stream>>>(
        seasInit, (long)1024 * FCIN, 1, FCIN, emb_dec_w, XD, 1024);

    // decoder self Fourier attention
    gemm_tok64(stream, XD, 64, dec_self_w + 0 * 4096, dec_self_b + 0 * 64, T1, 0);
    gemm_rfft(stream, T1, Gf, QF);
    mode_mix_kernel<<<32768 / 256, 256, 0, stream>>>(
        QF, dec_four_wr, dec_four_wi, OW, 1.0f);
    gemm_irfft(stream, OW, Gi, T2);
    gemm_tok64(stream, T2, 64, dec_self_w + 3 * 4096, dec_self_b + 3 * 64, T1, 0);
    decomp_kernel<<<FB * FD, 1024, 0, stream>>>(
        XD, T1, FD, 1024, (long)1024 * FD, 1, FD,
        decomp_w + 5 * 4, decomp_b + 5 * 4, XD, TSUM, 0, nullptr);

    // decoder cross Fourier attention
    gemm_tok64(stream, XD, 64, dec_cross_w + 0 * 4096, dec_cross_b + 0 * 64, T1, 0);
    gemm_rfft(stream, T1, Gf, QF);
    gemm_tok64(stream, HENC, 64, dec_cross_w + 1 * 4096, dec_cross_b + 1 * 64, T1, 0);
    gemm_rfft(stream, T1, Gf, KF);
    cross_qk_kernel<<<(FB * FH * 64 * 64) / 256, 256, 0, stream>>>(QF, KF, QKr, QKi);
    cross_qkv_kernel<<<32768 / 256, 256, 0, stream>>>(QKr, QKi, KF, QKV);
    mode_mix_kernel<<<32768 / 256, 256, 0, stream>>>(
        QKV, dec_cross_wr, dec_cross_wi, OW, 1.0f / 4096.0f);  // /(D*D)
    gemm_irfft(stream, OW, Gi, T2);
    gemm_tok64(stream, T2, 64, dec_cross_w + 3 * 4096, dec_cross_b + 3 * 64, T1, 0);
    decomp_kernel<<<FB * FD, 1024, 0, stream>>>(
        XD, T1, FD, 1024, (long)1024 * FD, 1, FD,
        decomp_w + 6 * 4, decomp_b + 6 * 4, XD, TSUM, 1, nullptr);

    // decoder FFN
    wmma_gemm_kernel<128><<<dim3(FTOK / 128, 2), 256, 0, stream>>>(
        XD, dec_c1, nullptr, FFNb, FTOK, 64, 256, 1, 64, 0, 1, 256, 0, 1);
    gemm_tok64(stream, FFNb, 256, dec_c2, nullptr, T1, 0);
    decomp_kernel<<<FB * FD, 1024, 0, stream>>>(
        XD, T1, FD, 1024, (long)1024 * FD, 1, FD,
        decomp_w + 7 * 4, decomp_b + 7 * 4, XD, TSUM, 1, nullptr);

    // trend assembly + final projection
    trend_conv_kernel<<<FB * 1024 / 256, 256, 0, stream>>>(
        TSUM, dec_trend_w, trendBase, trendFin);
    myln_pass1<<<FTOK / 256, 256, 0, stream>>>(XD, dec_ln_g, dec_ln_b, T2);
    colmean_sub_kernel<<<FB * FD, 256, 0, stream>>>(T2, 1024);
    final_out_kernel<<<(FB * FPRED) / 256, 256, 0, stream>>>(
        T2, dec_proj_w, dec_proj_b, trendFin, out);
}
